// ArchetipesNetwork_91113436217991
// MI455X (gfx1250) — matrix operand order repaired and checked by reading, not compiled, not run on hardware
//
#include <hip/hip_runtime.h>
#include <hip/hip_bf16.h>
#include <math.h>

// ---------------------------------------------------------------------------
// ArchetipesNetwork recurrent scan for MI455X (gfx1250, wave32, WMMA).
//
//  * conn[i,o]*scaling[i] folded into wm -> 4096x4096 matrix Afb, stored as
//    split bf16 (hi + residual lo) pre-swizzled into
//    A-fragment order with contiguous k = 16*(lane/16) + s. hi+lo WMMA products give
//    ~fp32-accurate matvecs while streaming bf16 from L2 (67 MB < 192 MB L2).
//  * W_rec -> same split/swizzle (block-diagonal, 16 modules of 256x256).
//  * inproj[t,r] = bias[r] + (mask ⊙ W_in) @ x[t] precomputed for all t with
//    full-N split-bf16 WMMA tiles (real GEMM, 4096x256x128).
//  * 256 sequential step kernels: 256 WGs x 8 waves. hy is converted once per
//    step into bf16 hi/lo LDS arrays in B-fragment order, so each K=32 chunk
//    is: 2 coalesced 32B global loads (A hi/lo) + 2 32B LDS reads + 3 WMMA.
//    Every C column replicates the matvec result (B columns identical), so no
//    predication is needed and EXEC stays all-ones around every WMMA.
//
// Workspace: 75.6 MB with split planes; falls back to hi-only (39.9 MB) if
// ws_size is smaller.
// ---------------------------------------------------------------------------

typedef __attribute__((ext_vector_type(16))) __bf16 v16bf;
typedef __attribute__((ext_vector_type(8)))  float  v8f;

#define DT_C    0.042f
#define GAMMA_C 2.7f
#define EPS_C   4.7f

constexpr int Mm = 16, Hh = 256, Ii = 128, Tt = 256;
constexpr int Rr = Mm * Hh;              // 4096 state rows
constexpr int RT = Rr / 16;              // 256 row tiles
constexpr int KC = Rr / 32;              // 128 k-chunks of the big matvec

// Workspace offsets (bytes). hi-only layout is a prefix of the split layout.
constexpr size_t OFF_SCALING = 0;                                    // 16 f32
constexpr size_t OFF_HY      = 256;                                  // 2*4096 f32
constexpr size_t OFF_HZ      = OFF_HY + 2 * Rr * sizeof(float);
constexpr size_t OFF_INPROJ  = OFF_HZ + 2 * Rr * sizeof(float);      // 4096*256 f32
constexpr size_t OFF_AFB_HI  = OFF_INPROJ + (size_t)Rr * Tt * sizeof(float);
constexpr size_t OFF_WREC_HI = OFF_AFB_HI + (size_t)Rr * Rr * 2;     // bf16
constexpr size_t OFF_AFB_LO  = OFF_WREC_HI + (size_t)Rr * Hh * 2;    // bf16
constexpr size_t OFF_WREC_LO = OFF_AFB_LO + (size_t)Rr * Rr * 2;     // bf16
constexpr size_t TOTAL_SPLIT = OFF_WREC_LO + (size_t)Rr * Hh * 2;

// A-fragment K map for 16-bit A (16x32): half h of lane -> K index.
__device__ __forceinline__ int akmap(int h, int hi) {
  return h + (hi ? 16 : 0);
}
// B-fragment K map for 16-bit B (32x16): half h of lane -> K (row) index.
__device__ __forceinline__ int bkmap(int h, int hi) {
  return h + (hi ? 16 : 0);
}

__device__ __forceinline__ void split_bf16(float v, __bf16& h, __bf16& l) {
  h = (__bf16)v;
  l = (__bf16)(v - (float)h);
}

// --- prep: scaling[i] = 1/max(1, sum_o conn[i,o]); zero state buffers ------
__global__ void prep_scale_zero(const float* __restrict__ conn,
                                float* __restrict__ scaling,
                                float* __restrict__ hy,
                                float* __restrict__ hz) {
  int tid = blockIdx.x * blockDim.x + threadIdx.x;
  if (blockIdx.x == 0 && threadIdx.x < Mm) {
    float s = 0.0f;
    for (int o = 0; o < Mm; ++o) s += conn[threadIdx.x * Mm + o];
    scaling[threadIdx.x] = 1.0f / fmaxf(s, 1.0f);
  }
  for (int i = tid; i < 2 * Rr; i += gridDim.x * blockDim.x) {
    hy[i] = 0.0f;
    hz[i] = 0.0f;
  }
}

// --- build Afb (split bf16, A-fragment-swizzled) ---------------------------
// gid = (rt*128 + kc)*32 + lane ; each thread writes its 16 half-elements.
__global__ void build_afb(const float* __restrict__ wm,
                          const float* __restrict__ conn,
                          const float* __restrict__ scaling,
                          __bf16* __restrict__ afb_hi,
                          __bf16* __restrict__ afb_lo,
                          int split) {
  int gid  = blockIdx.x * 256 + threadIdx.x;    // 0 .. 1048575
  int lane = gid & 31;
  int pair = gid >> 5;                           // rt*128 + kc
  int kc   = pair & (KC - 1);
  int rt   = pair >> 7;
  int hi   = (lane >= 16);
  int row  = rt * 16 + (lane & 15);
  int i    = row >> 8;
  int hh   = row & 255;
  int kbase = kc * 32;
  int o    = kbase >> 8;                         // uniform per chunk
  float cs = conn[i * Mm + o] * scaling[i];
  const float* wrow = wm + (((size_t)(i * Mm + o) * Hh + hh) * Hh);
  __bf16* dh = afb_hi + (size_t)gid * 16;
  __bf16* dl = afb_lo + (size_t)gid * 16;
#pragma unroll
  for (int h = 0; h < 16; ++h) {
    int g = (kbase + akmap(h, hi)) & 255;
    float v = cs * wrow[g];
    __bf16 vh, vl;
    split_bf16(v, vh, vl);
    dh[h] = vh;
    if (split) dl[h] = vl;
  }
}

// --- build W_rec (split bf16, A-fragment-swizzled) -------------------------
__global__ void build_wrec(const float* __restrict__ wrec,
                           __bf16* __restrict__ out_hi,
                           __bf16* __restrict__ out_lo,
                           int split) {
  int gid  = blockIdx.x * 256 + threadIdx.x;    // 0 .. 65535
  int lane = gid & 31;
  int pair = gid >> 5;                           // rt*8 + kc
  int kc   = pair & 7;
  int rt   = pair >> 3;
  int hi   = (lane >= 16);
  int row  = rt * 16 + (lane & 15);
  int m    = row >> 8;
  int hh   = row & 255;
  const float* wr = wrec + ((size_t)(m * Hh + hh)) * Hh;
  __bf16* dh = out_hi + (size_t)gid * 16;
  __bf16* dl = out_lo + (size_t)gid * 16;
#pragma unroll
  for (int h = 0; h < 16; ++h) {
    int g = kc * 32 + akmap(h, hi);
    __bf16 vh, vl;
    split_bf16(wr[g], vh, vl);
    dh[h] = vh;
    if (split) dl[h] = vl;
  }
}

// --- input projection GEMM: inproj[t,r] = bias[r] + (mask⊙W_in) @ x[t] -----
// One wave per 16x16 (row x time) tile, K = 128 in 4 chunks of 32,
// split-bf16 (3 WMMA per chunk) for ~fp32 accuracy.
__global__ void __launch_bounds__(256)
inproj_kernel(const float* __restrict__ x,
              const float* __restrict__ win,
              const float* __restrict__ mask,
              const float* __restrict__ bias,
              float* __restrict__ inproj) {
  int wave = threadIdx.x >> 5, lane = threadIdx.x & 31;
  int wgid = blockIdx.x * 8 + wave;              // 0 .. 4095 tiles
  int tt   = wgid & 15;
  int rt   = wgid >> 4;
  int hi   = (lane >= 16);
  int row  = rt * 16 + (lane & 15);
  int m    = row >> 8;
  int hh   = row & 255;
  float mk = mask[m];
  int tcol = tt * 16 + (lane & 15);

  v8f acc;
#pragma unroll
  for (int j = 0; j < 8; ++j) acc[j] = 0.0f;

  for (int kc = 0; kc < 4; ++kc) {
    v16bf ah, al, bh, bl;
#pragma unroll
    for (int h = 0; h < 16; ++h) {
      int ia = kc * 32 + akmap(h, hi);
      float av = win[((size_t)(m * Hh + hh)) * Ii + ia] * mk;
      __bf16 t0, t1;
      split_bf16(av, t0, t1);
      ah[h] = t0; al[h] = t1;
      int ib = kc * 32 + bkmap(h, hi);
      float bv = x[(size_t)tcol * Ii + ib];
      split_bf16(bv, t0, t1);
      bh[h] = t0; bl[h] = t1;
    }
    acc = __builtin_amdgcn_wmma_f32_16x16x32_bf16(false, ah, false, bh,
                                                  (short)0, acc, false, false);
    acc = __builtin_amdgcn_wmma_f32_16x16x32_bf16(false, ah, false, bl,
                                                  (short)0, acc, false, false);
    acc = __builtin_amdgcn_wmma_f32_16x16x32_bf16(false, al, false, bh,
                                                  (short)0, acc, false, false);
  }
#pragma unroll
  for (int j = 0; j < 8; ++j) {
    int r = rt * 16 + j + (hi ? 8 : 0);
    inproj[(size_t)tcol * Rr + r] = acc[j] + bias[r];
  }
}

// --- one recurrence step ---------------------------------------------------
// grid 256 (row tiles) x 256 threads (8 waves). Wave w covers k-chunks
// [16w, 16w+16) of the feedback matvec + chunk w of W_rec.
__global__ void __launch_bounds__(256)
step_kernel(const __bf16* __restrict__ afb_hi,
            const __bf16* __restrict__ afb_lo,
            const __bf16* __restrict__ wrec_hi,
            const __bf16* __restrict__ wrec_lo,
            const float* __restrict__ inproj,
            const float* __restrict__ hy_in,
            const float* __restrict__ hz_in,
            float* __restrict__ hy_out,
            float* __restrict__ hz_out,
            float* __restrict__ out_state,
            float* __restrict__ out_fb,
            int t, int split) {
  // hy in bf16 hi/lo, laid out so a chunk's B fragment is one contiguous
  // 32B run per half-wave: K = h + 16*laneHi  ->  plain linear order.
  __shared__ __align__(32) __bf16 hyh_s[Rr];
  __shared__ __align__(32) __bf16 hyl_s[Rr];
  __shared__ float red_fb[8][16];
  __shared__ float red_rec[8][16];

  int tid = threadIdx.x, lane = tid & 31, wave = tid >> 5;
  int rt = blockIdx.x;

  for (int i = tid; i < Rr; i += 256) {
    __bf16 vh, vl;
    split_bf16(hy_in[i], vh, vl);
    hyh_s[i] = vh;
    hyl_s[i] = vl;
  }
  __syncthreads();

  int hi   = (lane >= 16);
  int off16 = hi ? 16 : 0;

  v8f acc;
#pragma unroll
  for (int j = 0; j < 8; ++j) acc[j] = 0.0f;

  // feedback matvec: 16 chunks of K=32 per wave, hi/lo split products.
  const size_t fragbase = (size_t)((rt * KC + wave * 16) * 32 + lane) * 16;
#pragma unroll 2
  for (int c = 0; c < 16; ++c) {
    int kc = wave * 16 + c;
    const __bf16* pah = afb_hi + fragbase + (size_t)c * 32 * 16;
    // prefetch next chunk's fragments while this chunk computes
    __builtin_prefetch(pah + 32 * 16, 0, 1);
    v16bf ahi = *(const v16bf*)pah;
    v16bf bhi = *(const v16bf*)&hyh_s[kc * 32 + off16];
    acc = __builtin_amdgcn_wmma_f32_16x16x32_bf16(false, ahi, false, bhi,
                                                  (short)0, acc, false, false);
    if (split) {
      const __bf16* pal = afb_lo + fragbase + (size_t)c * 32 * 16;
      __builtin_prefetch(pal + 32 * 16, 0, 1);
      v16bf alo = *(const v16bf*)pal;
      v16bf blo = *(const v16bf*)&hyl_s[kc * 32 + off16];
      acc = __builtin_amdgcn_wmma_f32_16x16x32_bf16(
          false, ahi, false, blo, (short)0, acc, false, false);
      acc = __builtin_amdgcn_wmma_f32_16x16x32_bf16(
          false, alo, false, bhi, (short)0, acc, false, false);
    }
  }

  // recurrent (block-diagonal) matvec: 1 chunk per wave.
  v8f acc2;
#pragma unroll
  for (int j = 0; j < 8; ++j) acc2[j] = 0.0f;
  {
    int m = rt >> 4;
    int kb = m * Hh + wave * 32 + off16;
    size_t wb = (size_t)((rt * 8 + wave) * 32 + lane) * 16;
    v16bf ahi = *(const v16bf*)(wrec_hi + wb);
    v16bf bhi = *(const v16bf*)&hyh_s[kb];
    acc2 = __builtin_amdgcn_wmma_f32_16x16x32_bf16(false, ahi, false, bhi,
                                                   (short)0, acc2, false, false);
    if (split) {
      v16bf alo = *(const v16bf*)(wrec_lo + wb);
      v16bf blo = *(const v16bf*)&hyl_s[kb];
      acc2 = __builtin_amdgcn_wmma_f32_16x16x32_bf16(
          false, ahi, false, blo, (short)0, acc2, false, false);
      acc2 = __builtin_amdgcn_wmma_f32_16x16x32_bf16(
          false, alo, false, bhi, (short)0, acc2, false, false);
    }
  }

  // Every C column replicates the matvec; take column 0 from lanes 0 / 16.
  if ((lane & 15) == 0) {
    int off = hi ? 8 : 0;
#pragma unroll
    for (int j = 0; j < 8; ++j) {
      red_fb[wave][off + j]  = acc[j];
      red_rec[wave][off + j] = acc2[j];
    }
  }
  __syncthreads();

  if (tid < 16) {
    float fb = 0.0f, rc = 0.0f;
#pragma unroll
    for (int w = 0; w < 8; ++w) {
      fb += red_fb[w][tid];
      rc += red_rec[w][tid];
    }
    int r = rt * 16 + tid;
    float hy = hy_in[r];
    float hz = hz_in[r];
    float pre = inproj[(size_t)t * Rr + r] + fb + rc;
    float hzn = hz + DT_C * (tanhf(pre) - GAMMA_C * hy - EPS_C * hz);
    float hyn = hy + DT_C * hzn;
    hy_out[r] = hyn;
    hz_out[r] = hzn;
    int m = r >> 8, h = r & 255;
    out_state[(size_t)t * (Mm * 2 * Hh) + m * (2 * Hh) + h]      = hyn;
    out_state[(size_t)t * (Mm * 2 * Hh) + m * (2 * Hh) + Hh + h] = hzn;
    out_fb[(size_t)t * Rr + r] = fb;
  }
}

extern "C" void kernel_launch(void* const* d_in, const int* in_sizes, int n_in,
                              void* d_out, int out_size, void* d_ws,
                              size_t ws_size, hipStream_t stream) {
  const float* x    = (const float*)d_in[0];  // (T, I)
  const float* wm   = (const float*)d_in[1];  // (M, M, H, H)
  const float* conn = (const float*)d_in[2];  // (M, M)
  const float* mask = (const float*)d_in[3];  // (M,)
  const float* win  = (const float*)d_in[4];  // (M, H, I)
  const float* wrec = (const float*)d_in[5];  // (M, H, H)
  const float* bias = (const float*)d_in[6];  // (M, H)
  (void)in_sizes; (void)n_in; (void)out_size;

  const int split = (ws_size >= TOTAL_SPLIT) ? 1 : 0;

  char* ws = (char*)d_ws;
  float*  scaling = (float*)(ws + OFF_SCALING);
  float*  hy      = (float*)(ws + OFF_HY);
  float*  hz      = (float*)(ws + OFF_HZ);
  float*  inproj  = (float*)(ws + OFF_INPROJ);
  __bf16* afb_hi  = (__bf16*)(ws + OFF_AFB_HI);
  __bf16* wrs_hi  = (__bf16*)(ws + OFF_WREC_HI);
  __bf16* afb_lo  = split ? (__bf16*)(ws + OFF_AFB_LO)  : afb_hi;
  __bf16* wrs_lo  = split ? (__bf16*)(ws + OFF_WREC_LO) : wrs_hi;

  prep_scale_zero<<<64, 256, 0, stream>>>(conn, scaling, hy, hz);
  build_afb<<<4096, 256, 0, stream>>>(wm, conn, scaling, afb_hi, afb_lo, split);
  build_wrec<<<256, 256, 0, stream>>>(wrec, wrs_hi, wrs_lo, split);
  inproj_kernel<<<512, 256, 0, stream>>>(x, win, mask, bias, inproj);

  float* out_state = (float*)d_out;                        // (T, M, 2, H)
  float* out_fb    = out_state + (size_t)Tt * Mm * 2 * Hh; // (T, M, H)

  for (int t = 0; t < Tt; ++t) {
    const float* hyi = hy + (t & 1) * Rr;
    float*       hyo = hy + ((t + 1) & 1) * Rr;
    const float* hzi = hz + (t & 1) * Rr;
    float*       hzo = hz + ((t + 1) & 1) * Rr;
    step_kernel<<<RT, 256, 0, stream>>>(afb_hi, afb_lo, wrs_hi, wrs_lo, inproj,
                                        hyi, hzi, hyo, hzo, out_state, out_fb,
                                        t, split);
  }
}
